// BaseModel_91173565759958
// MI455X (gfx1250) — matrix operand order repaired and checked by reading, not compiled, not run on hardware
//
#include <hip/hip_runtime.h>
#include <hip/hip_bf16.h>
#include <cstdint>
#include <cstddef>

typedef __attribute__((ext_vector_type(16))) _Float16 v16h;
typedef __attribute__((ext_vector_type(8)))  _Float16 v8h;
typedef __attribute__((ext_vector_type(8)))  float    v8f;

#define NB_TOT 16000   // N*B
#define NNODE  500
#define BATCH  32
#define TSTEPS 32
#define HDIM   64

__device__ __forceinline__ v16h pack16(v8h a, v8h b) {
  return __builtin_shufflevector(a, b, 0,1,2,3,4,5,6,7,8,9,10,11,12,13,14,15);
}
__device__ __forceinline__ v8f wmma_f16(v16h a, v16h b, v8f c) {
  // (neg_a, A, neg_b, B, c_mod, C, reuse_a, reuse_b)
  return __builtin_amdgcn_wmma_f32_16x16x32_f16(false, a, false, b, (short)0, c, false, false);
}
// fast gate nonlinearities: v_exp_f32 + v_rcp_f32, no IEEE divide sequence
__device__ __forceinline__ float fast_sigmoid(float x) {
  return __builtin_amdgcn_rcpf(1.0f + __expf(-x));
}
__device__ __forceinline__ float fast_tanh(float x) {
  // tanh(x) = 1 - 2/(1+exp(2x)); rcp(inf)=0 gives correct +/-1 saturation
  return fmaf(-2.0f, __builtin_amdgcn_rcpf(1.0f + __expf(2.0f * x)), 1.0f);
}

// ---------------------------------------------------------------------------
// GRU layer. One wave owns 16 sequences (WMMA M=16). Per step:
//   gates[16x192] = x_t @ Wih^T + h @ Whh^T (+ biases), via v_wmma_f32_16x16x32_f16
// B-fragments pre-swizzled in LDS (compiler hoists the loop-invariant ones to
// VGPRs); h kept in registers (f32, C layout) and mirrored in LDS as f16
// row-major to source next step's A fragments.
// LAYER==1: x = f32 [NB,T,15] (K zero-padded to 32), writes h1 as f16.
// LAYER==2: x = f16 [NB,T,64], writes h2 as f32.
// ---------------------------------------------------------------------------
template<int LAYER>
__global__ __launch_bounds__(128) void gru_kernel(
    const void* __restrict__ xin,
    const float* __restrict__ Wih, const float* __restrict__ Whh,
    const float* __restrict__ bih, const float* __restrict__ bhh,
    _Float16* __restrict__ out16, float* __restrict__ out32)
{
  constexpr int DIN = (LAYER == 1) ? 15 : 64;
  constexpr int XC  = (LAYER == 1) ? 1  : 2;   // K-chunks of 32 for the x GEMM

  __shared__ _Float16 lds_bw[2][12][32][16];   // Whh B-fragments (24 KB)
  __shared__ _Float16 lds_bx[XC][12][32][16];  // Wih B-fragments (12/24 KB)
  __shared__ _Float16 lds_h [4][16][64];       // per-wave h mirror (8 KB)

  const int tid  = threadIdx.x;
  const int wave = tid >> 5;
  const int lane = tid & 31;
  const int m    = lane & 15;   // row for A, column-in-tile for B/C
  const int hi   = lane >> 4;

  // Stage recurrent-weight B fragments: B[k][n] = Whh[n][k]
  for (int x = tid; x < 2*12*32*16; x += 128) {
    int e = x & 15, l = (x >> 4) & 31, f = x >> 9;
    int nt = f % 12, kc = f / 12;
    int n = nt*16 + (l & 15);
    int k = kc*32 + ((e >> 3) << 4) + ((l >> 4) << 3) + (e & 7);
    lds_bw[kc][nt][l][e] = (_Float16)Whh[n*HDIM + k];
  }
  // Stage input-weight B fragments (zero-pad K beyond DIN)
  for (int x = tid; x < XC*12*32*16; x += 128) {
    int e = x & 15, l = (x >> 4) & 31, f = x >> 9;
    int nt = f % 12, kc = f / 12;
    int n = nt*16 + (l & 15);
    int k = kc*32 + ((e >> 3) << 4) + ((l >> 4) << 3) + (e & 7);
    lds_bx[kc][nt][l][e] = (k < DIN) ? (_Float16)Wih[n*DIN + k] : (_Float16)0.0f;
  }
  for (int x = tid; x < 4*16*64; x += 128) ((_Float16*)lds_h)[x] = (_Float16)0.0f;
  __syncthreads();

  // Per-lane gate biases (column depends only on lane&15)
  float bias_r[4], bias_z[4], bias_xn[4], bias_hn[4];
  for (int nt = 0; nt < 4; ++nt) {
    int c0 = nt*16 + m;
    bias_r [nt] = bih[c0]       + bhh[c0];
    bias_z [nt] = bih[64 + c0]  + bhh[64 + c0];
    bias_xn[nt] = bih[128 + c0];
    bias_hn[nt] = bhh[128 + c0];
  }

  const int seqBase = blockIdx.x * 64 + wave * 16;
  _Float16* myh = &lds_h[wave][0][0];

  float hreg[4][8];                       // h in C layout: row=r+8*hi, col=nt*16+m
  for (int nt = 0; nt < 4; ++nt)
    for (int r = 0; r < 8; ++r) hreg[nt][r] = 0.0f;

  for (int t = 0; t < TSTEPS; ++t) {
    // ---- A fragments for x_t ----
    v16h ax[XC];
    if constexpr (LAYER == 1) {
      const float* xr = ((const float*)xin) + ((size_t)(seqBase + m)*TSTEPS + t)*15;
      v16h a;
      for (int e = 0; e < 16; ++e) a[e] = (_Float16)0.0f;       // K 15..31 padded
      for (int e = 0; e < 8; ++e) { int k = 8*hi + e; if (k < 15) a[e] = (_Float16)xr[k]; }
      ax[0] = a;
      if (t + 1 < TSTEPS) __builtin_prefetch(xr + 15, 0, 1);
    } else {
      const _Float16* xr = ((const _Float16*)xin) + ((size_t)(seqBase + m)*TSTEPS + t)*HDIM;
      for (int kc = 0; kc < XC; ++kc) {
        v8h g0 = *(const v8h*)(xr + kc*32 + 8*hi);
        v8h g1 = *(const v8h*)(xr + kc*32 + 16 + 8*hi);
        ax[kc] = pack16(g0, g1);
      }
      if (t + 1 < TSTEPS) __builtin_prefetch(xr + HDIM, 0, 1);
    }
    // ---- A fragments for h_{t-1} (LDS f16 mirror) ----
    v16h ah[2];
    for (int kc = 0; kc < 2; ++kc) {
      v8h g0 = *(const v8h*)(myh + m*64 + kc*32 + 8*hi);
      v8h g1 = *(const v8h*)(myh + m*64 + kc*32 + 16 + 8*hi);
      ah[kc] = pack16(g0, g1);
    }

    // ---- gate GEMMs + lane-local GRU update, 4 column-groups ----
    for (int nt = 0; nt < 4; ++nt) {
      v8f ar, az, axn, ahn;
      for (int r = 0; r < 8; ++r) {
        ar[r] = bias_r[nt]; az[r] = bias_z[nt];
        axn[r] = bias_xn[nt]; ahn[r] = bias_hn[nt];
      }
      for (int kc = 0; kc < XC; ++kc) {
        ar  = wmma_f16(ax[kc], pack16(*(const v8h*)&lds_bx[kc][nt    ][lane][0], *(const v8h*)&lds_bx[kc][nt    ][lane][8]), ar);
        az  = wmma_f16(ax[kc], pack16(*(const v8h*)&lds_bx[kc][nt + 4][lane][0], *(const v8h*)&lds_bx[kc][nt + 4][lane][8]), az);
        axn = wmma_f16(ax[kc], pack16(*(const v8h*)&lds_bx[kc][nt + 8][lane][0], *(const v8h*)&lds_bx[kc][nt + 8][lane][8]), axn);
      }
      for (int kc = 0; kc < 2; ++kc) {
        ar  = wmma_f16(ah[kc], pack16(*(const v8h*)&lds_bw[kc][nt    ][lane][0], *(const v8h*)&lds_bw[kc][nt    ][lane][8]), ar);
        az  = wmma_f16(ah[kc], pack16(*(const v8h*)&lds_bw[kc][nt + 4][lane][0], *(const v8h*)&lds_bw[kc][nt + 4][lane][8]), az);
        ahn = wmma_f16(ah[kc], pack16(*(const v8h*)&lds_bw[kc][nt + 8][lane][0], *(const v8h*)&lds_bw[kc][nt + 8][lane][8]), ahn);
      }
      // r_j, z_j, n_j for column j live in the same lane: pointwise update
      for (int r = 0; r < 8; ++r) {
        float rr = fast_sigmoid(ar[r]);
        float zz = fast_sigmoid(az[r]);
        float nn = fast_tanh(axn[r] + rr * ahn[r]);
        float hn = (1.0f - zz) * nn + zz * hreg[nt][r];
        hreg[nt][r] = hn;
        myh[(r + 8*hi)*64 + nt*16 + m] = (_Float16)hn;
      }
    }

    // ---- emit h_t (coalesced from LDS mirror, same wave => DS in-order) ----
    if constexpr (LAYER == 1) {
      for (int i = 0; i < 4; ++i) {
        int g = lane*4 + i, row = g >> 3, grp = g & 7;
        v8h v = *(const v8h*)(myh + row*64 + grp*8);
        *(v8h*)(out16 + ((size_t)(seqBase + row)*TSTEPS + t)*HDIM + grp*8) = v;
      }
    } else {
      for (int i = 0; i < 4; ++i) {
        int g = lane*4 + i, row = g >> 3, grp = g & 7;
        v8h v = *(const v8h*)(myh + row*64 + grp*8);
        float* dst = out32 + ((size_t)(seqBase + row)*TSTEPS + t)*HDIM + grp*8;
        float4 f0 = make_float4((float)v[0], (float)v[1], (float)v[2], (float)v[3]);
        float4 f1 = make_float4((float)v[4], (float)v[5], (float)v[6], (float)v[7]);
        *(float4*)dst = f0; *(float4*)(dst + 4) = f1;
      }
    }
  }
}

// ---------------------------------------------------------------------------
// Temporal attention: one wave per sequence. lane = timestep.
// ---------------------------------------------------------------------------
__global__ __launch_bounds__(32) void attn_kernel(
    const float* __restrict__ h2, const float* __restrict__ attnW,
    const float* __restrict__ attnB, float* __restrict__ Ai)
{
  const int seq  = blockIdx.x;
  const int lane = threadIdx.x;
  const float* hrow = h2 + (size_t)seq * TSTEPS * HDIM + (size_t)lane * HDIM;
  float s = attnB[0];
  for (int j = 0; j < HDIM; ++j) s += hrow[j] * attnW[j];
  s = fast_tanh(s);
  float mx = s;
  for (int off = 16; off; off >>= 1) mx = fmaxf(mx, __shfl_xor(mx, off, 32));
  float e = __expf(s - mx);
  float den = e;
  for (int off = 16; off; off >>= 1) den += __shfl_xor(den, off, 32);
  float w = e * __builtin_amdgcn_rcpf(den);
  float a0 = 0.0f, a1 = 0.0f;
  for (int tt = 0; tt < TSTEPS; ++tt) {
    float wt = __shfl(w, tt, 32);
    const float* hr = h2 + (size_t)seq * TSTEPS * HDIM + (size_t)tt * HDIM;
    a0 += wt * hr[lane];
    a1 += wt * hr[lane + 32];
  }
  float* out = Ai + (size_t)seq * HDIM;
  out[lane] = a0; out[lane + 32] = a1;
}

// ---------------------------------------------------------------------------
// GAT projections: sq = Ai@Ww^T + b -> s_q/s_k scores; proj = Ai@W1^T + b1.
// One thread per (node,batch) row.
// ---------------------------------------------------------------------------
__global__ __launch_bounds__(256) void gat_setup_kernel(
    const float* __restrict__ Ai, const float* __restrict__ Ww,
    const float* __restrict__ Wb, const float* __restrict__ u,
    const float* __restrict__ W1w, const float* __restrict__ W1b,
    float* __restrict__ proj, float* __restrict__ s_q, float* __restrict__ s_k)
{
  int row = blockIdx.x * blockDim.x + threadIdx.x;
  if (row >= NB_TOT) return;
  float a[HDIM];
  const float* ar = Ai + (size_t)row * HDIM;
  for (int j = 0; j < HDIM; ++j) a[j] = ar[j];
  float sqv = 0.0f, skv = 0.0f;
  for (int i = 0; i < HDIM; ++i) {
    float v = Wb[i];
    for (int k = 0; k < HDIM; ++k) v += a[k] * Ww[i*HDIM + k];
    sqv += v * u[i];
    skv += v * u[HDIM + i];
  }
  s_q[row] = sqv; s_k[row] = skv;
  for (int i = 0; i < HDIM; ++i) {
    float v = W1b[i];
    for (int k = 0; k < HDIM; ++k) v += a[k] * W1w[i*HDIM + k];
    proj[(size_t)row * HDIM + i] = v;
  }
}

// ---------------------------------------------------------------------------
// GAT aggregation: one block per (q,b); thread j accumulates feature j.
// proj (4 MB) stays resident in L2 across all 500 q-blocks per b.
// ---------------------------------------------------------------------------
__global__ __launch_bounds__(64) void gat_agg_kernel(
    const float* __restrict__ proj, const float* __restrict__ s_q,
    const float* __restrict__ s_k, float* __restrict__ out)
{
  const int qb = blockIdx.x;       // q*BATCH + b
  const int b  = qb & (BATCH - 1);
  const int j  = threadIdx.x;
  const float sq = s_q[qb];
  float acc = 0.0f, den = 0.0f;
  for (int k = 0; k < NNODE; ++k) {
    float sc = sq + s_k[k*BATCH + b];
    sc = (sc > 0.0f) ? sc : 0.01f * sc;          // leaky_relu(0.01)
    float c = __expf(sc);
    den += c;
    acc += c * proj[((size_t)k*BATCH + b)*HDIM + j];
  }
  float g = acc * __builtin_amdgcn_rcpf(den);
  out[(size_t)qb * HDIM + j] = fmaxf(g, 0.0f);
}

// ---------------------------------------------------------------------------
extern "C" void kernel_launch(void* const* d_in, const int* in_sizes, int n_in,
                              void* d_out, int out_size, void* d_ws, size_t ws_size,
                              hipStream_t stream) {
  (void)in_sizes; (void)n_in; (void)out_size; (void)ws_size;
  const float* raw      = (const float*)d_in[0];
  const float* g1_Wih   = (const float*)d_in[1];
  const float* g1_Whh   = (const float*)d_in[2];
  const float* g1_bih   = (const float*)d_in[3];
  const float* g1_bhh   = (const float*)d_in[4];
  const float* g2_Wih   = (const float*)d_in[5];
  const float* g2_Whh   = (const float*)d_in[6];
  const float* g2_bih   = (const float*)d_in[7];
  const float* g2_bhh   = (const float*)d_in[8];
  const float* attn_W   = (const float*)d_in[9];
  const float* attn_b   = (const float*)d_in[10];
  const float* gat_W_w  = (const float*)d_in[11];
  const float* gat_W_b  = (const float*)d_in[12];
  const float* gat_u    = (const float*)d_in[13];
  const float* gat_W1_w = (const float*)d_in[14];
  const float* gat_W1_b = (const float*)d_in[15];

  char* ws = (char*)d_ws;
  const size_t h1_bytes   = (size_t)NB_TOT * TSTEPS * HDIM * sizeof(_Float16); //  65.5 MB
  const size_t h2_bytes   = (size_t)NB_TOT * TSTEPS * HDIM * sizeof(float);    // 131.1 MB
  const size_t ai_bytes   = (size_t)NB_TOT * HDIM * sizeof(float);             //   4.1 MB
  const size_t proj_bytes = ai_bytes;
  const size_t sq_bytes   = (size_t)NB_TOT * sizeof(float);

  _Float16* h1  = (_Float16*)(ws);
  float*    h2  = (float*)(ws + h1_bytes);
  float*    Ai  = (float*)(ws + h1_bytes + h2_bytes);
  float*    pr  = (float*)(ws + h1_bytes + h2_bytes + ai_bytes);
  float*    s_q = (float*)(ws + h1_bytes + h2_bytes + ai_bytes + proj_bytes);
  float*    s_k = (float*)(ws + h1_bytes + h2_bytes + ai_bytes + proj_bytes + sq_bytes);

  // 64 sequences per block (4 waves x 16), 250 blocks
  gru_kernel<1><<<NB_TOT/64, 128, 0, stream>>>((const void*)raw, g1_Wih, g1_Whh, g1_bih, g1_bhh, h1, nullptr);
  gru_kernel<2><<<NB_TOT/64, 128, 0, stream>>>((const void*)h1,  g2_Wih, g2_Whh, g2_bih, g2_bhh, nullptr, h2);
  attn_kernel<<<NB_TOT, 32, 0, stream>>>(h2, attn_W, attn_b, Ai);
  gat_setup_kernel<<<(NB_TOT + 255)/256, 256, 0, stream>>>(Ai, gat_W_w, gat_W_b, gat_u, gat_W1_w, gat_W1_b, pr, s_q, s_k);
  gat_agg_kernel<<<NB_TOT, 64, 0, stream>>>(pr, s_q, s_k, (float*)d_out);
}
